// GatedDeltaNetLayer_5102421147671
// MI455X (gfx1250) — matrix operand order repaired and checked by reading, not compiled, not run on hardware
//
#include <hip/hip_runtime.h>
#include <hip/hip_bf16.h>

typedef __bf16 bf16;
typedef __attribute__((ext_vector_type(16))) __bf16 v16bf;
typedef __attribute__((ext_vector_type(8)))  float  v8f;
typedef int vsi4 __attribute__((__vector_size__(4 * sizeof(int))));

static constexpr int B_ = 2, T_ = 4096, D_ = 1024, H_ = 16, K_ = 64, V_ = 64;
static constexpr int MROWS = B_ * T_;       // 8192
static constexpr int N1  = 2080;            // [k(1024) | v(1024) | beta(16) | g(16)]
static constexpr int N1P = 2112;            // padded to multiple of 64
static constexpr int MARKER_ = 50251;
static constexpr int CHUNK = 48;            // scan staging chunk (double buffered)

union ABu { v16bf v; uint4 u[2]; };

// ---- CDNA5 async global->LDS staging (ASYNCcnt-tracked DMA, no VGPR round trip)
#if __has_builtin(__builtin_amdgcn_global_load_async_to_lds_b128) && \
    __has_builtin(__builtin_amdgcn_global_load_async_to_lds_b32)
#define ASYNC_STAGE 1
// signature (probe-confirmed from diagnostics): (AS1 typed ptr, AS3 typed ptr, imm offset, imm cpol)
#define ASYNC_B128(gsrc, ldst)                                                  \
  __builtin_amdgcn_global_load_async_to_lds_b128(                               \
      (__attribute__((address_space(1))) vsi4*)(void*)(const void*)(gsrc),      \
      (__attribute__((address_space(3))) vsi4*)(void*)(ldst), 0, 0)
#define ASYNC_B32(gsrc, ldst)                                                   \
  __builtin_amdgcn_global_load_async_to_lds_b32(                                \
      (__attribute__((address_space(1))) int*)(void*)(const void*)(gsrc),       \
      (__attribute__((address_space(3))) int*)(void*)(ldst), 0, 0)
#if __has_builtin(__builtin_amdgcn_s_wait_asynccnt)
#define WAIT_ASYNC(n) __builtin_amdgcn_s_wait_asynccnt(n)
#else
#define WAIT_ASYNC(n) asm volatile("s_wait_asynccnt %0" :: "i"(n) : "memory")
#endif
#else
#define ASYNC_STAGE 0
#endif

__device__ __forceinline__ void bf16split(float x, bf16& hi, bf16& lo) {
  hi = (bf16)x;
  lo = (bf16)(x - (float)hi);
}

// ---------------------------------------------------------------------------
// Marker slot computation: slot[row] = (cumsum(mask)-1) % 64 if marker else -1
// ---------------------------------------------------------------------------
__global__ void slot_kernel(const int* __restrict__ ids, int* __restrict__ slot) {
  int b = threadIdx.x;
  if (b >= B_) return;
  int cnt = 0;
  for (int t = 0; t < T_; ++t) {
    int m = (ids[b * T_ + t] == MARKER_);
    cnt += m;
    slot[b * T_ + t] = m ? ((cnt - 1) & 63) : -1;
  }
}

// ---------------------------------------------------------------------------
// RMSNorm + bf16 hi/lo split of activations
// ---------------------------------------------------------------------------
__global__ __launch_bounds__(256)
void rmsnorm_kernel(const float* __restrict__ x, const float* __restrict__ nw,
                    bf16* __restrict__ XHI, bf16* __restrict__ XLO) {
  const int row = blockIdx.x;
  const int tid = threadIdx.x;
  const float4* xr = (const float4*)(x + (size_t)row * D_);
  float4 xv = xr[tid];
  float ss = xv.x * xv.x + xv.y * xv.y + xv.z * xv.z + xv.w * xv.w;
  __shared__ float red[256];
  red[tid] = ss;
  __syncthreads();
  for (int off = 128; off > 0; off >>= 1) {
    if (tid < off) red[tid] += red[tid + off];
    __syncthreads();
  }
  const float inv = 1.0f / sqrtf(red[0] * (1.0f / (float)D_) + 1e-6f);
  const float4* wr = (const float4*)nw;
  float4 wv = wr[tid];
  float o0 = xv.x * inv * wv.x;
  float o1 = xv.y * inv * wv.y;
  float o2 = xv.z * inv * wv.z;
  float o3 = xv.w * inv * wv.w;
  bf16* oh = XHI + (size_t)row * D_ + tid * 4;
  bf16* ol = XLO + (size_t)row * D_ + tid * 4;
  bf16 h, l;
  bf16split(o0, h, l); oh[0] = h; ol[0] = l;
  bf16split(o1, h, l); oh[1] = h; ol[1] = l;
  bf16split(o2, h, l); oh[2] = h; ol[2] = l;
  bf16split(o3, h, l); oh[3] = h; ol[3] = l;
}

// ---------------------------------------------------------------------------
// Weight packing into WMMA B-operand lane layout (bf16 hi/lo split).
// B tile (32k x 16n): lane L supplies column n = L&15, k = ktile*32 + 16*(L>>4) + j
// ---------------------------------------------------------------------------
__global__ __launch_bounds__(256)
void pack_w1(const float* __restrict__ kw, const float* __restrict__ vw,
             const float* __restrict__ bw, const float* __restrict__ gw,
             bf16* __restrict__ Whi, bf16* __restrict__ Wlo) {
  const int NT = N1P / 16;                       // 132 column tiles
  int tid = blockIdx.x * 256 + threadIdx.x;
  if (tid >= 32 * NT * 32) return;               // ktiles * ntiles * lanes
  int lane = tid & 31;
  int tile = tid >> 5;
  int kt = tile / NT, nt = tile % NT;
  int n  = nt * 16 + (lane & 15);
  int kb = kt * 32 + 16 * (lane >> 4);
  size_t base = ((size_t)tile * 32 + lane) * 16;
  #pragma unroll
  for (int j = 0; j < 16; ++j) {
    int k = kb + j;
    float wv2;
    if (n < 1024)      wv2 = kw[(size_t)k * 1024 + n];
    else if (n < 2048) wv2 = vw[(size_t)k * 1024 + (n - 1024)];
    else if (n < 2064) wv2 = bw[(size_t)k * 16 + (n - 2048)];
    else if (n < 2080) wv2 = gw[(size_t)k * 16 + (n - 2064)];
    else               wv2 = 0.0f;
    bf16 h, l; bf16split(wv2, h, l);
    Whi[base + j] = h; Wlo[base + j] = l;
  }
}

__global__ __launch_bounds__(256)
void pack_w2(const float* __restrict__ ow, bf16* __restrict__ Whi, bf16* __restrict__ Wlo) {
  const int NT = 64;                             // 1024/16 column tiles
  int tid = blockIdx.x * 256 + threadIdx.x;
  if (tid >= 32 * NT * 32) return;
  int lane = tid & 31;
  int tile = tid >> 5;
  int kt = tile / NT, nt = tile % NT;
  int n  = nt * 16 + (lane & 15);
  int kb = kt * 32 + 16 * (lane >> 4);
  size_t base = ((size_t)tile * 32 + lane) * 16;
  #pragma unroll
  for (int j = 0; j < 16; ++j) {
    int k = kb + j;
    float wv2 = ow[(size_t)k * 1024 + n];
    bf16 h, l; bf16split(wv2, h, l);
    Whi[base + j] = h; Wlo[base + j] = l;
  }
}

// ---------------------------------------------------------------------------
// Split-bf16 WMMA GEMM: Y = Ahi*Bhi + Ahi*Blo + Alo*Bhi (+ residual)
// Block = 8 waves; each wave computes a 16x64 tile; block tile = 128 x 64.
// ---------------------------------------------------------------------------
__global__ __launch_bounds__(256)
void gemm_bf16x3(const bf16* __restrict__ Ahi, const bf16* __restrict__ Alo,
                 const bf16* __restrict__ Bhi, const bf16* __restrict__ Blo,
                 float* __restrict__ Y, const float* __restrict__ residual,
                 int Npad, int Kdim) {
  const int l     = threadIdx.x & 31;
  const int w     = threadIdx.x >> 5;
  const int lrow  = l & 15;
  const int lhalf = l >> 4;
  const int row0  = blockIdx.y * 128 + w * 16;
  const int col0  = blockIdx.x * 64;
  const int Ntiles = Npad >> 4;

  v8f acc[4];
  #pragma unroll
  for (int c = 0; c < 4; ++c)
    acc[c] = (v8f){0.f, 0.f, 0.f, 0.f, 0.f, 0.f, 0.f, 0.f};

  for (int k0 = 0; k0 < Kdim; k0 += 32) {
    ABu ah, al;
    // A operand: lane<16 -> K {k0..k0+15}; lane>=16 -> K {k0+16..k0+31}
    const uint4* pah = (const uint4*)(Ahi + (size_t)(row0 + lrow) * Kdim + k0 + lhalf * 16);
    const uint4* pal = (const uint4*)(Alo + (size_t)(row0 + lrow) * Kdim + k0 + lhalf * 16);
    ah.u[0] = pah[0]; ah.u[1] = pah[1];
    al.u[0] = pal[0]; al.u[1] = pal[1];
    const int tbase = (k0 >> 5) * Ntiles + (col0 >> 4);
    #pragma unroll
    for (int c = 0; c < 4; ++c) {
      ABu bh, bl;
      const uint4* pbh = (const uint4*)(Bhi + ((size_t)(tbase + c) * 32 + l) * 16);
      const uint4* pbl = (const uint4*)(Blo + ((size_t)(tbase + c) * 32 + l) * 16);
      bh.u[0] = pbh[0]; bh.u[1] = pbh[1];
      bl.u[0] = pbl[0]; bl.u[1] = pbl[1];
      acc[c] = __builtin_amdgcn_wmma_f32_16x16x32_bf16(false, ah.v, false, bh.v, (short)0, acc[c], false, false);
      acc[c] = __builtin_amdgcn_wmma_f32_16x16x32_bf16(false, ah.v, false, bl.v, (short)0, acc[c], false, false);
      acc[c] = __builtin_amdgcn_wmma_f32_16x16x32_bf16(false, al.v, false, bh.v, (short)0, acc[c], false, false);
    }
  }
  #pragma unroll
  for (int c = 0; c < 4; ++c) {
    #pragma unroll
    for (int r = 0; r < 8; ++r) {
      int rr = row0 + lhalf * 8 + r;       // C layout: vgpr r -> M=r (lanes 0-15), M=r+8 (16-31)
      int cc = col0 + c * 16 + lrow;
      float val = acc[c][r];
      if (residual) val += residual[(size_t)rr * Npad + cc];
      Y[(size_t)rr * Npad + cc] = val;
    }
  }
}

// ---------------------------------------------------------------------------
// In-place postprocess of GEMM-1 output Y [8192 x 2112]
// ---------------------------------------------------------------------------
__global__ __launch_bounds__(512)
void postprocess_kernel(float* __restrict__ Y, const int* __restrict__ slot,
                        const float* __restrict__ key_bank,
                        const float* __restrict__ beta_b, const float* __restrict__ g_b) {
  const int row  = blockIdx.x;
  const int lane = threadIdx.x & 31;
  const int h    = threadIdx.x >> 5;      // 0..15
  const int sl   = slot[row];
  const bool mk  = (sl >= 0);
  float* yk = Y + (size_t)row * N1P + h * K_;
  const int e0 = lane * 2;
  float a0, a1;
  if (mk) {
    const float* kb = key_bank + ((size_t)h * 64 + sl) * 64;
    a0 = kb[e0]; a1 = kb[e0 + 1];
  } else {
    a0 = yk[e0]; a1 = yk[e0 + 1];
  }
  float ss = a0 * a0 + a1 * a1;
  #pragma unroll
  for (int m = 16; m > 0; m >>= 1) ss += __shfl_xor(ss, m, 32);
  float inv = 1.0f / fmaxf(sqrtf(ss), 1e-12f);
  yk[e0]     = a0 * inv;
  yk[e0 + 1] = a1 * inv;

  if (threadIdx.x < 32) {                 // wave 0 also handles beta/g columns
    float* yr = Y + (size_t)row * N1P + 2048;
    float scale = mk ? 1.0f : 0.1f;
    if (lane < 16) {
      float z = yr[lane] + beta_b[lane];
      yr[lane] = scale / (1.0f + expf(-z));
    } else {
      int hh = lane - 16;
      float z = yr[16 + hh] + g_b[hh];
      yr[16 + hh] = 1.0f / (1.0f + expf(-z));
    }
  }
}

// ---------------------------------------------------------------------------
// Gated delta-rule scan. One block per (b,h); 64 threads; thread v owns S[:,v].
// ||k||==1  =>  o = g*pred + beta*(v - pred).
// k/v/beta/g double-buffered in LDS via CDNA5 async global->LDS DMA
// (ASYNCcnt), so staging latency hides behind the previous chunk's compute.
// ---------------------------------------------------------------------------
__global__ __launch_bounds__(64)
void scan_kernel(const float* __restrict__ Y, const float* __restrict__ init_state,
                 float* __restrict__ out_state,
                 bf16* __restrict__ OHI, bf16* __restrict__ OLO) {
  const int b = blockIdx.x >> 4;
  const int h = blockIdx.x & 15;
  const int v = threadIdx.x;

  float S[64];
  #pragma unroll
  for (int k = 0; k < 64; ++k)
    S[k] = init_state[(((size_t)(b * H_ + h)) * 64 + k) * 64 + v];

  __shared__ float kbuf[2][CHUNK * 64];
  __shared__ float vbuf[2][CHUNK * 64];
  __shared__ float bbuf[2][CHUNK];
  __shared__ float gbuf[2][CHUNK];

  // --- stage one chunk (34 async instructions per active thread) ---
  auto stage = [&](int c0, int n, int buf) {
    const int s = threadIdx.x;
    if (s < n) {
      const int t = c0 + s;
      const float* kp = Y + (size_t)(b * T_ + t) * N1P + h * K_;
      const float* vp = Y + (size_t)(b * T_ + t + 1) * N1P + 1024 + h * V_;
      float* kd = &kbuf[buf][s * 64];
      float* vd = &vbuf[buf][s * 64];
#if ASYNC_STAGE
      #pragma unroll
      for (int j = 0; j < 16; ++j) {
        ASYNC_B128(kp + 4 * j, kd + 4 * j);
        ASYNC_B128(vp + 4 * j, vd + 4 * j);
      }
      ASYNC_B32(Y + (size_t)(b * T_ + t) * N1P + 2048 + h, &bbuf[buf][s]);
      ASYNC_B32(Y + (size_t)(b * T_ + t) * N1P + 2064 + h, &gbuf[buf][s]);
#else
      const float4* kp4 = (const float4*)kp;
      const float4* vp4 = (const float4*)vp;
      float4* kd4 = (float4*)kd;
      float4* vd4 = (float4*)vd;
      #pragma unroll
      for (int j = 0; j < 16; ++j) { kd4[j] = kp4[j]; vd4[j] = vp4[j]; }
      bbuf[buf][s] = Y[(size_t)(b * T_ + t) * N1P + 2048 + h];
      gbuf[buf][s] = Y[(size_t)(b * T_ + t) * N1P + 2064 + h];
#endif
    }
  };

  stage(0, (T_ - 1 < CHUNK) ? (T_ - 1) : CHUNK, 0);

  int buf = 0;
  for (int c0 = 0; c0 < T_ - 1; c0 += CHUNK) {
    const int n = (T_ - 1 - c0 < CHUNK) ? (T_ - 1 - c0) : CHUNK;
    const int c1 = c0 + CHUNK;
    const bool has_next = c1 < T_ - 1;
    if (has_next) {
      stage(c1, (T_ - 1 - c1 < CHUNK) ? (T_ - 1 - c1) : CHUNK, buf ^ 1);
    }
#if ASYNC_STAGE
    if (has_next) { WAIT_ASYNC(34); } else { WAIT_ASYNC(0); }
#endif
    __syncthreads();

    for (int s = 0; s < n; ++s) {
      float4 kk[16];
      const float4* ks = (const float4*)&kbuf[buf][s * 64];
      #pragma unroll
      for (int j = 0; j < 16; ++j) kk[j] = ks[j];
      const float vt = vbuf[buf][s * 64 + v];
      const float bt = bbuf[buf][s];
      const float gt = gbuf[buf][s];
      float p0 = 0.f, p1 = 0.f, p2 = 0.f, p3 = 0.f;
      #pragma unroll
      for (int j = 0; j < 16; ++j) {
        p0 = fmaf(kk[j].x, S[4 * j + 0], p0);
        p1 = fmaf(kk[j].y, S[4 * j + 1], p1);
        p2 = fmaf(kk[j].z, S[4 * j + 2], p2);
        p3 = fmaf(kk[j].w, S[4 * j + 3], p3);
      }
      const float pred = (p0 + p1) + (p2 + p3);
      const float cc = bt * (vt - pred);
      const float o  = fmaf(gt, pred, cc);     // k^T S_new with ||k||=1
      #pragma unroll
      for (int j = 0; j < 16; ++j) {
        S[4 * j + 0] = fmaf(gt, S[4 * j + 0], cc * kk[j].x);
        S[4 * j + 1] = fmaf(gt, S[4 * j + 1], cc * kk[j].y);
        S[4 * j + 2] = fmaf(gt, S[4 * j + 2], cc * kk[j].z);
        S[4 * j + 3] = fmaf(gt, S[4 * j + 3], cc * kk[j].w);
      }
      const size_t oi = (size_t)(b * T_ + c0 + s + 1) * (size_t)(H_ * V_) + h * V_ + v;
      bf16 hh, ll; bf16split(o, hh, ll);
      OHI[oi] = hh; OLO[oi] = ll;
    }
    __syncthreads();
    buf ^= 1;
  }
  #pragma unroll
  for (int k = 0; k < 64; ++k)
    out_state[(((size_t)(b * H_ + h)) * 64 + k) * 64 + v] = S[k];
}

// ---------------------------------------------------------------------------
extern "C" void kernel_launch(void* const* d_in, const int* in_sizes, int n_in,
                              void* d_out, int out_size, void* d_ws, size_t ws_size,
                              hipStream_t stream) {
  (void)in_sizes; (void)n_in; (void)out_size; (void)ws_size;

  const float* x          = (const float*)d_in[0];
  const int*   ids        = (const int*)  d_in[1];
  const float* init_state = (const float*)d_in[2];
  const float* key_bank   = (const float*)d_in[3];
  const float* norm_w     = (const float*)d_in[4];
  const float* k_w        = (const float*)d_in[5];
  const float* v_w        = (const float*)d_in[6];
  const float* o_w        = (const float*)d_in[7];
  const float* beta_w     = (const float*)d_in[8];
  const float* beta_b     = (const float*)d_in[9];
  const float* g_w        = (const float*)d_in[10];
  const float* g_b        = (const float*)d_in[11];

  float* out       = (float*)d_out;
  float* out_state = out + (size_t)MROWS * D_;

  // --- workspace carve ---
  char* p = (char*)d_ws;
  auto carve = [&](size_t bytes) -> void* {
    void* r = (void*)p;
    p += (bytes + 255) & ~(size_t)255;
    return r;
  };
  bf16*  XHI  = (bf16*) carve((size_t)MROWS * D_ * sizeof(bf16));
  bf16*  XLO  = (bf16*) carve((size_t)MROWS * D_ * sizeof(bf16));
  bf16*  W1H  = (bf16*) carve((size_t)32 * (N1P / 16) * 32 * 16 * sizeof(bf16));
  bf16*  W1L  = (bf16*) carve((size_t)32 * (N1P / 16) * 32 * 16 * sizeof(bf16));
  bf16*  W2H  = (bf16*) carve((size_t)32 * 64 * 32 * 16 * sizeof(bf16));
  bf16*  W2L  = (bf16*) carve((size_t)32 * 64 * 32 * 16 * sizeof(bf16));
  float* Yb   = (float*)carve((size_t)MROWS * N1P * sizeof(float));
  bf16*  OHIb = (bf16*) carve((size_t)MROWS * D_ * sizeof(bf16));
  bf16*  OLOb = (bf16*) carve((size_t)MROWS * D_ * sizeof(bf16));
  int*   SLOT = (int*)  carve((size_t)MROWS * sizeof(int));

  // 1. marker slots (sequential cumsum per batch)
  slot_kernel<<<1, 32, 0, stream>>>(ids, SLOT);

  // 2. rmsnorm + bf16 hi/lo activations
  rmsnorm_kernel<<<MROWS, 256, 0, stream>>>(x, norm_w, XHI, XLO);

  // 3. pack weights into WMMA lane layout (hi/lo)
  pack_w1<<<(32 * (N1P / 16) * 32 + 255) / 256, 256, 0, stream>>>(k_w, v_w, beta_w, g_w, W1H, W1L);
  pack_w2<<<(32 * 64 * 32 + 255) / 256, 256, 0, stream>>>(o_w, W2H, W2L);

  // 4. GEMM-1: Y = x_norm @ [k_w | v_w | beta_w | g_w]
  gemm_bf16x3<<<dim3(N1P / 64, MROWS / 128), 256, 0, stream>>>(XHI, XLO, W1H, W1L, Yb, nullptr, N1P, D_);

  // 5. marker substitution + k-normalize + sigmoid(beta,g), in place on Y
  postprocess_kernel<<<MROWS, 512, 0, stream>>>(Yb, SLOT, key_bank, beta_b, g_b);

  // 6. zero O (rows t=0 per batch must stay zero), then sequential delta-rule scan
  (void)hipMemsetAsync(OHIb, 0, (size_t)MROWS * D_ * sizeof(bf16), stream);
  (void)hipMemsetAsync(OLOb, 0, (size_t)MROWS * D_ * sizeof(bf16), stream);
  scan_kernel<<<B_ * H_, 64, 0, stream>>>(Yb, init_state, out_state, OHIb, OLOb);

  // 7. GEMM-2: output = x + O @ o_w
  gemm_bf16x3<<<dim3(D_ / 64, MROWS / 128), 256, 0, stream>>>(OHIb, OLOb, W2H, W2L, out, x, D_, D_);
}
